// MultiheadAttention_80444737454400
// MI455X (gfx1250) — matrix operand order repaired and checked by reading, not compiled, not run on hardware
//
#include <hip/hip_runtime.h>
#include <stdint.h>

// ---------------------------------------------------------------------------
// MI455X (gfx1250) fused multi-head attention, wave32 + WMMA bf16.
// HBM-bound on attn_bias (805 MB) -> flash attention streams bias once into
// the WMMA C accumulator; K/V tiles double-buffered with async global->LDS
// copies (ASYNCcnt) so their HBM latency hides behind bias loads + WMMA.
// ---------------------------------------------------------------------------

typedef __attribute__((ext_vector_type(16))) __bf16 v16bf;
typedef __attribute__((ext_vector_type(8)))  float  v8f;

union Frag { v16bf v; uint32_t u[8]; };

#if defined(__gfx1250__) && __has_builtin(__builtin_amdgcn_global_load_async_to_lds_b128)
#define HAVE_ASYNC_LDS 1
#endif

// Exact pointee type from the builtin's prototype: int __vector_size__(16)
typedef int v4i __attribute__((__vector_size__(4 * sizeof(int))));
typedef __attribute__((address_space(1))) v4i* g4p;   // global (__device__) AS
typedef __attribute__((address_space(3))) v4i* l4p;   // LDS (__shared__) AS

__device__ __forceinline__ void wait_async0() {
#ifdef HAVE_ASYNC_LDS
#if __has_builtin(__builtin_amdgcn_s_wait_asynccnt)
  __builtin_amdgcn_s_wait_asynccnt(0);
#else
  asm volatile("s_wait_asynccnt 0x0" ::: "memory");
#endif
#endif
}

__device__ __forceinline__ uint16_t f2bf(float f) {
  uint32_t u = __builtin_bit_cast(uint32_t, f);
  u += 0x7fffu + ((u >> 16) & 1u);          // round-to-nearest-even
  return (uint16_t)(u >> 16);
}

constexpr int T = 2048, B = 4, E = 768, H = 12, D = 64;
constexpr int M = T * B;                    // 8192 GEMM rows (t*B + b)
constexpr float SCALE = 0.125f;             // D^-0.5

// A-frag k order chosen here (contiguous, same as B-frags): VGPR j holds K = kbaseA, kbaseA+1 (K = 2j + 16*half)
__device__ __forceinline__ int kbaseA(int j, int half) {
  return 2 * j + 16 * half;
}

__device__ __forceinline__ v8f wmma_bf16(const Frag& a, const Frag& b, v8f c) {
  return __builtin_amdgcn_wmma_f32_16x16x32_bf16(false, a.v, false, b.v,
                                                 (short)0, c, false, false);
}

// ------------------------------ fp32 -> bf16 -------------------------------
__global__ void cvt_bf16(const float* __restrict__ src,
                         uint16_t* __restrict__ dst, int n) {
  for (int i = blockIdx.x * blockDim.x + threadIdx.x; i < n;
       i += gridDim.x * blockDim.x)
    dst[i] = f2bf(src[i]);
}

// --------------------- shared WMMA GEMM tile (16 x 64) ---------------------
// A: [M, E] bf16 row-major, Bw: [N, E] bf16 row-major (i.e. computes A @ Bw^T)
__device__ __forceinline__ void gemm_tile(const uint16_t* __restrict__ A,
                                          const uint16_t* __restrict__ Bw,
                                          int mrow0, int ncol0, v8f c[4]) {
  const int lane = threadIdx.x & 31, l16 = lane & 15, half = lane >> 4;
  const uint32_t* A32 = (const uint32_t*)A;
  const uint32_t* B32 = (const uint32_t*)Bw;
  const int arow = mrow0 + l16;
  for (int k0 = 0; k0 < E; k0 += 32) {
    Frag a;
#pragma unroll
    for (int j = 0; j < 8; ++j)
      a.u[j] = A32[(arow * E + k0 + kbaseA(j, half)) >> 1];
#pragma unroll
    for (int n = 0; n < 4; ++n) {
      Frag b;
      const int ncol = ncol0 + 16 * n + l16;          // B-frag: lane = column
#pragma unroll
      for (int j = 0; j < 8; ++j)                     // VGPR j: K = 2j + 16*half
        b.u[j] = B32[(ncol * E + k0 + 2 * j + 16 * half) >> 1];
      c[n] = wmma_bf16(a, b, c[n]);
    }
  }
}

// ------------------- QKV projection, scatter to [B,H,T,D] ------------------
__global__ __launch_bounds__(128)
void qkv_gemm(const uint16_t* __restrict__ xb, const uint16_t* __restrict__ wb,
              const float* __restrict__ bias, uint16_t* __restrict__ out,
              float scale) {
  const int wave = threadIdx.x >> 5;
  const int mrow0 = blockIdx.x * 64 + wave * 16;
  const int ncol0 = blockIdx.y * 64;
  v8f c[4];
#pragma unroll
  for (int n = 0; n < 4; ++n)
#pragma unroll
    for (int r = 0; r < 8; ++r) c[n][r] = 0.f;
  gemm_tile(xb, wb, mrow0, ncol0, c);
  const int lane = threadIdx.x & 31, l16 = lane & 15, half = lane >> 4;
#pragma unroll
  for (int n = 0; n < 4; ++n) {
    const int col = ncol0 + 16 * n + l16;
    const int h = col >> 6, d = col & 63;
    const float bb = bias[col];
#pragma unroll
    for (int r = 0; r < 8; ++r) {
      const int m = mrow0 + r + 8 * half;             // C-frag: M = r + 8*half
      const int t = m >> 2, b = m & 3;                // m = t*B + b
      out[((b * H + h) * T + t) * D + d] = f2bf((c[n][r] + bb) * scale);
    }
  }
}

// ------------------------------ flash attention ----------------------------
__global__ __launch_bounds__(256)
void flash_attn(const uint16_t* __restrict__ Qb, const uint16_t* __restrict__ Kb,
                const uint16_t* __restrict__ Vb, const float* __restrict__ bias,
                const float* __restrict__ mask,
                const unsigned char* __restrict__ pad,
                uint16_t* __restrict__ Ob) {
  __shared__ __attribute__((aligned(16))) uint16_t Ks[2][64][72]; // [key][d]
  __shared__ __attribute__((aligned(16))) uint16_t Vs[2][64][72]; // [key][d]
  __shared__ __attribute__((aligned(16))) uint16_t Ps[8][16][72]; // per-wave P

  const int bh = blockIdx.y;
  const int b = bh / H, h = bh % H;
  const int wave = threadIdx.x >> 5;
  const int lane = threadIdx.x & 31, l16 = lane & 15, half = lane >> 4;
  const int qrow0 = blockIdx.x * 128 + wave * 16;

  const uint32_t* Q32 = (const uint32_t*)Qb;

  // ---- async (double-buffered) K/V tile staging: straight 8KB copies ----
  auto stage = [&](int s0, int buf) {
    const size_t gelem = (size_t)(bh * T + s0) * D;   // bf16 element offset
#pragma unroll
    for (int i = 0; i < 2; ++i) {
      const int c = (int)threadIdx.x + 256 * i;       // 512 chunks of 16B
      const int key = c >> 3, dcol = (c & 7) * 8;
#ifdef HAVE_ASYNC_LDS
      __builtin_amdgcn_global_load_async_to_lds_b128(
          (g4p)(Kb + gelem + (size_t)c * 8),
          (l4p)&Ks[buf][key][dcol], 0, 0);
      __builtin_amdgcn_global_load_async_to_lds_b128(
          (g4p)(Vb + gelem + (size_t)c * 8),
          (l4p)&Vs[buf][key][dcol], 0, 0);
#else
      *(uint4*)&Ks[buf][key][dcol] = *(const uint4*)(Kb + gelem + (size_t)c * 8);
      *(uint4*)&Vs[buf][key][dcol] = *(const uint4*)(Vb + gelem + (size_t)c * 8);
#endif
    }
  };

  // Q A-fragments, d = 0..31 and 32..63 (registers for the whole pass)
  Frag qa0, qa1;
  {
    const int arow = bh * T + qrow0 + l16;
#pragma unroll
    for (int j = 0; j < 8; ++j) {
      const int ka = kbaseA(j, half);
      qa0.u[j] = Q32[(arow * D + ka) >> 1];
      qa1.u[j] = Q32[(arow * D + 32 + ka) >> 1];
    }
  }

  v8f o[4];
  float mrow[8], lsum[8];
#pragma unroll
  for (int n = 0; n < 4; ++n)
#pragma unroll
    for (int r = 0; r < 8; ++r) o[n][r] = 0.f;
#pragma unroll
  for (int r = 0; r < 8; ++r) { mrow[r] = -1e30f; lsum[r] = 0.f; }

  stage(0, 0);                                        // prologue fill

  for (int it = 0; it < T / 64; ++it) {
    const int s0 = it * 64, buf = it & 1;
    wait_async0();          // own async copies for tile `it` have landed
    __syncthreads();        // ... and everyone else's too
    if (it + 1 < T / 64) stage(s0 + 64, buf ^ 1);     // overlap next tile

    // ---- scores: C preloaded with bias+mask, then Q @ K^T ----
    v8f sc[4];
#pragma unroll
    for (int n = 0; n < 4; ++n) {
      const int scol = s0 + 16 * n + l16;
#pragma unroll
      for (int r = 0; r < 8; ++r) {
        const int trow = qrow0 + r + 8 * half;
        sc[n][r] = bias[(bh * T + trow) * T + scol] + mask[trow * T + scol];
      }
      if (s0 + 64 < T)   // stream next bias tile (global_prefetch_b8)
        __builtin_prefetch(&bias[(bh * T + qrow0 + 8 * half) * T + scol + 64], 0, 1);
      Frag kb0, kb1;
      const int krow = 16 * n + l16;                  // B-frag: lane = key col
#pragma unroll
      for (int j = 0; j < 8; ++j) {
        const int kk = 2 * j + 16 * half;
        kb0.u[j] = *(const uint32_t*)&Ks[buf][krow][kk];
        kb1.u[j] = *(const uint32_t*)&Ks[buf][krow][32 + kk];
      }
      sc[n] = wmma_bf16(qa0, kb0, sc[n]);
      sc[n] = wmma_bf16(qa1, kb1, sc[n]);
      if (pad[b * T + scol]) {                        // column-wise -inf
#pragma unroll
        for (int r = 0; r < 8; ++r) sc[n][r] = -1e30f;
      }
    }

    // ---- online softmax (row stats across half-wave: width-16 shfl) ----
    float mt[8];
#pragma unroll
    for (int r = 0; r < 8; ++r)
      mt[r] = fmaxf(fmaxf(sc[0][r], sc[1][r]), fmaxf(sc[2][r], sc[3][r]));
#pragma unroll
    for (int d = 1; d < 16; d <<= 1)
#pragma unroll
      for (int r = 0; r < 8; ++r) mt[r] = fmaxf(mt[r], __shfl_xor(mt[r], d, 16));

    float corr[8], ls[8];
#pragma unroll
    for (int r = 0; r < 8; ++r) {
      const float mn = fmaxf(mrow[r], mt[r]);
      corr[r] = __expf(mrow[r] - mn);
      mrow[r] = mn;
      ls[r] = 0.f;
    }
#pragma unroll
    for (int n = 0; n < 4; ++n)
#pragma unroll
      for (int r = 0; r < 8; ++r) {
        const float p = __expf(sc[n][r] - mrow[r]);
        ls[r] += p;
        Ps[wave][r + 8 * half][16 * n + l16] = f2bf(p);  // C-layout -> LDS
      }
#pragma unroll
    for (int d = 1; d < 16; d <<= 1)
#pragma unroll
      for (int r = 0; r < 8; ++r) ls[r] += __shfl_xor(ls[r], d, 16);
#pragma unroll
    for (int r = 0; r < 8; ++r) lsum[r] = lsum[r] * corr[r] + ls[r];
#pragma unroll
    for (int n = 0; n < 4; ++n)
#pragma unroll
      for (int r = 0; r < 8; ++r) o[n][r] *= corr[r];

    // ---- P (A-layout from LDS) @ V (B-layout: 2 x u16 pack per VGPR) ----
    Frag pa0, pa1;
#pragma unroll
    for (int j = 0; j < 8; ++j) {
      const int ka = kbaseA(j, half);
      pa0.u[j] = *(const uint32_t*)&Ps[wave][l16][ka];
      pa1.u[j] = *(const uint32_t*)&Ps[wave][l16][32 + ka];
    }
#pragma unroll
    for (int n = 0; n < 4; ++n) {
      Frag vb0, vb1;
      const int dcol = 16 * n + l16;                  // B-frag: lane = d col
#pragma unroll
      for (int j = 0; j < 8; ++j) {
        const int k0 = 2 * j + 16 * half;             // VGPR j: keys k0, k0+1
        vb0.u[j] = (uint32_t)Vs[buf][k0][dcol] |
                   ((uint32_t)Vs[buf][k0 + 1][dcol] << 16);
        vb1.u[j] = (uint32_t)Vs[buf][k0 + 32][dcol] |
                   ((uint32_t)Vs[buf][k0 + 33][dcol] << 16);
      }
      o[n] = wmma_bf16(pa0, vb0, o[n]);
      o[n] = wmma_bf16(pa1, vb1, o[n]);
    }
  }

  // ---- normalize, store O as bf16 rows (t*B + b) for the output GEMM ----
  float inv[8];
#pragma unroll
  for (int r = 0; r < 8; ++r) inv[r] = 1.f / lsum[r];
#pragma unroll
  for (int n = 0; n < 4; ++n)
#pragma unroll
    for (int r = 0; r < 8; ++r) {
      const int t = qrow0 + r + 8 * half;
      const int col = h * D + 16 * n + l16;
      Ob[(t * B + b) * E + col] = f2bf(o[n][r] * inv[r]);
    }
}

// --------------------------- output projection -----------------------------
__global__ __launch_bounds__(128)
void out_gemm(const uint16_t* __restrict__ Ob, const uint16_t* __restrict__ wob,
              const float* __restrict__ bo, float* __restrict__ out) {
  const int wave = threadIdx.x >> 5;
  const int mrow0 = blockIdx.x * 64 + wave * 16;
  const int ncol0 = blockIdx.y * 64;
  v8f c[4];
#pragma unroll
  for (int n = 0; n < 4; ++n)
#pragma unroll
    for (int r = 0; r < 8; ++r) c[n][r] = 0.f;
  gemm_tile(Ob, wob, mrow0, ncol0, c);
  const int lane = threadIdx.x & 31, l16 = lane & 15, half = lane >> 4;
#pragma unroll
  for (int n = 0; n < 4; ++n) {
    const int col = ncol0 + 16 * n + l16;
    const float bb = bo[col];
#pragma unroll
    for (int r = 0; r < 8; ++r) {
      const int m = mrow0 + r + 8 * half;
      out[m * E + col] = c[n][r] + bb;
    }
  }
}

// ------------------------------- launcher ----------------------------------
extern "C" void kernel_launch(void* const* d_in, const int* in_sizes, int n_in,
                              void* d_out, int out_size, void* d_ws,
                              size_t ws_size, hipStream_t stream) {
  (void)in_sizes; (void)n_in; (void)out_size; (void)ws_size;
  const float* query     = (const float*)d_in[0];
  const float* attn_bias = (const float*)d_in[1];
  const float* attn_mask = (const float*)d_in[2];
  const unsigned char* pad = (const unsigned char*)d_in[3];  // jax bool mask
  const float* wq = (const float*)d_in[4];
  const float* bq = (const float*)d_in[5];
  const float* wk = (const float*)d_in[6];
  const float* bk = (const float*)d_in[7];
  const float* wv = (const float*)d_in[8];
  const float* bv = (const float*)d_in[9];
  const float* wo = (const float*)d_in[10];
  const float* bo = (const float*)d_in[11];

  char* ws = (char*)d_ws;
  size_t off = 0;
  auto alloc = [&](size_t bytes) -> void* {
    void* p = ws + off;
    off = (off + bytes + 255) & ~(size_t)255;
    return p;
  };
  uint16_t* xb  = (uint16_t*)alloc((size_t)M * E * 2);
  uint16_t* wqb = (uint16_t*)alloc((size_t)E * E * 2);
  uint16_t* wkb = (uint16_t*)alloc((size_t)E * E * 2);
  uint16_t* wvb = (uint16_t*)alloc((size_t)E * E * 2);
  uint16_t* wob = (uint16_t*)alloc((size_t)E * E * 2);
  uint16_t* Qb  = (uint16_t*)alloc((size_t)B * H * T * D * 2);
  uint16_t* Kb  = (uint16_t*)alloc((size_t)B * H * T * D * 2);
  uint16_t* Vb  = (uint16_t*)alloc((size_t)B * H * T * D * 2);
  uint16_t* Ob  = (uint16_t*)alloc((size_t)M * E * 2);

  cvt_bf16<<<dim3(2048), 256, 0, stream>>>(query, xb, M * E);
  cvt_bf16<<<dim3(576), 256, 0, stream>>>(wq, wqb, E * E);
  cvt_bf16<<<dim3(576), 256, 0, stream>>>(wk, wkb, E * E);
  cvt_bf16<<<dim3(576), 256, 0, stream>>>(wv, wvb, E * E);
  cvt_bf16<<<dim3(576), 256, 0, stream>>>(wo, wob, E * E);

  qkv_gemm<<<dim3(M / 64, E / 64), 128, 0, stream>>>(xb, wqb, bq, Qb, SCALE);
  qkv_gemm<<<dim3(M / 64, E / 64), 128, 0, stream>>>(xb, wkb, bk, Kb, 1.0f);
  qkv_gemm<<<dim3(M / 64, E / 64), 128, 0, stream>>>(xb, wvb, bv, Vb, 1.0f);

  flash_attn<<<dim3(T / 128, B * H), 256, 0, stream>>>(
      Qb, Kb, Vb, attn_bias, attn_mask, pad, Ob);

  out_gemm<<<dim3(M / 64, E / 64), 128, 0, stream>>>(Ob, wob, bo, (float*)d_out);
}
